// BasicBlock_86036784873838
// MI455X (gfx1250) — matrix operand order repaired and checked by reading, not compiled, not run on hardware
//
#include <hip/hip_runtime.h>
#include <hip/hip_bf16.h>
#include <cstdint>
#include <cstddef>

typedef __attribute__((ext_vector_type(16))) __bf16 v16bf;
typedef __attribute__((ext_vector_type(8)))  float  v8f;
typedef __attribute__((ext_vector_type(16))) unsigned short v16u;

#define BB   16
#define CC   64
#define OO   64
#define HWD  56
#define HWHW (56*56)            // 3136
#define CK   576                // 64*9
#define NPIX (BB*OO*HWHW)       // 3,211,264
#define NRED (BB*HWHW)          // 50176 (BN reduction count)
#define EPSV 1e-5f
#define DELTA    (6.0f/15.0f)
#define INVDELTA (15.0f/6.0f)

// ---------------- K0: zero the BN-stat accumulators (256 floats) --------------
__global__ void k0_zero(float* __restrict__ s){
    if (threadIdx.x < 256) s[threadIdx.x] = 0.0f;
}

// ---------------- K1: adder layer 1 (VALU, LDS-tiled) + BN1 stats -------------
// Block = 256 threads, 8x8 output tile for one batch image, 4 o-stages of 16.
__global__ void k1_adder1(const float* __restrict__ x, const float* __restrict__ w1,
                          float* __restrict__ out1, float* __restrict__ gstats){
    __shared__ float wl[16*CK];     // 36,864 B : 16 output-channel filter slice
    __shared__ float xt[64*100];    // 25,600 B : 10x10 halo tile, 64 channels
    __shared__ float st[128];       //    512 B : per-channel sum / sumsq
    int tid = threadIdx.x;
    int bid = blockIdx.x;
    int b   = bid / 49, t49 = bid - b*49;
    int y0  = (t49 / 7) * 8, x0 = (t49 % 7) * 8;

    for (int i = tid; i < 6400; i += 256){
        int c  = i / 100, r = i - c*100;
        int iy = y0 + r/10 - 1, ix = x0 + (r%10) - 1;
        float v = 0.0f;
        if ((unsigned)iy < 56u && (unsigned)ix < 56u)
            v = x[((size_t)(b*CC + c))*HWHW + iy*HWD + ix];
        xt[i] = v;
    }
    if (tid < 128) st[tid] = 0.0f;

    int p = tid & 63, sub = tid >> 6;     // position in tile, o-subgroup
    int py = p >> 3,  px  = p & 7;

    for (int stage = 0; stage < 4; ++stage){
        __syncthreads();
        for (int i = tid; i < 16*CK; i += 256) wl[i] = w1[stage*16*CK + i];
        __syncthreads();

        float acc[4] = {0.f, 0.f, 0.f, 0.f};
        for (int c = 0; c < CC; ++c){
            #pragma unroll
            for (int kh = 0; kh < 3; ++kh){
                const float* xr = &xt[c*100 + (py+kh)*10 + px];
                float xa = xr[0], xb = xr[1], xc = xr[2];
                int wb = c*9 + kh*3;
                #pragma unroll
                for (int i = 0; i < 4; ++i){
                    const float* wr = &wl[(sub*4 + i)*CK + wb];
                    acc[i] += fabsf(xa - wr[0]) + fabsf(xb - wr[1]) + fabsf(xc - wr[2]);
                }
            }
        }
        #pragma unroll
        for (int i = 0; i < 4; ++i){
            int o = stage*16 + sub*4 + i;
            float v = -acc[i];
            out1[((size_t)(b*OO + o))*HWHW + (y0+py)*HWD + (x0+px)] = v;
            atomicAdd(&st[o],      v);
            atomicAdd(&st[64 + o], v*v);
        }
    }
    __syncthreads();
    if (tid < 128) atomicAdd(&gstats[tid], st[tid]);   // [0..63]=sum1 [64..127]=ssq1
}

// ---------------- K2: BN1 + ReLU + clip + 4-bit quantize -> u8 levels ---------
__global__ void k2_bnq(const float* __restrict__ out1, const float* __restrict__ gstats,
                       const float* __restrict__ s1, const float* __restrict__ b1,
                       unsigned char* __restrict__ act1q){
    int idx = blockIdx.x * 256 + threadIdx.x;
    if (idx >= NPIX) return;
    int o = (idx / HWHW) & 63;
    float mean = gstats[o]      * (1.0f/NRED);
    float var  = gstats[64 + o] * (1.0f/NRED) - mean*mean;
    float inv  = rsqrtf(var + EPSV);
    float y = (out1[idx] - mean) * inv * s1[o] + b1[o];
    y = fminf(fmaxf(y, 0.0f), 6.0f);
    act1q[idx] = (unsigned char)(int)rintf(y * INVDELTA);   // level 0..15
}

// ---------------- K3: build bf16 LUT T[o,ck,l]=|l*delta - w2| in B-frag layout -
// B-frag layout this code uses for V_WMMA_F32_16X16X32_BF16 (contiguous k, A matches):
//   frag index (kk, nt): lane = n + 16*hi holds K = 16*hi + j in half j,
//   where k = cl*16 + l, cl = ck&1 (two ck per 32-k step), j = l.
__global__ void k3_lut(const float* __restrict__ w2, unsigned short* __restrict__ Tf){
    int idx = blockIdx.x * 256 + threadIdx.x;
    if (idx >= OO*CK*16) return;
    int l  = idx & 15;
    int t  = idx >> 4;
    int ck = t % CK;
    int o  = t / CK;
    float val = fabsf((float)l * DELTA - w2[o*CK + ck]);
    unsigned u = __float_as_uint(val);                       // fp32 -> bf16 RNE
    unsigned short h = (unsigned short)((u + 0x7FFFu + ((u >> 16) & 1u)) >> 16);
    int kk = ck >> 1, cl = ck & 1, nt = o >> 4, n = o & 15;
    Tf[(((size_t)(kk*4 + nt))*32 + (n + 16*cl))*16 + l] = h;
}

// ---------------- K4: adder layer 2 as one-hot x LUT GEMM on WMMA -------------
// Block = 64 threads = 2 waves; block covers 4 M-tiles (64 positions).
// Each wave: 2 M-tiles x 4 N-tiles -> 8 WMMAs per k-step; one-hot A built in
// registers feeds 4 matrix ops, each B-frag feeds 2. B-frags are double-
// buffered across k-steps so load latency hides under the 8 in-flight WMMAs.
// __launch_bounds__(64,2): allow up to 512 VGPRs/wave -> no spills for the
// ~190-register working set (8 acc + 8 B-frag + 2 A-frag tiles).
// K = CK*16 = 9216 -> 288 steps of V_WMMA_F32_16X16X32_BF16.
__global__ void __launch_bounds__(64, 2)
k4_adder2_wmma(const unsigned char* __restrict__ act1q,
               const unsigned short* __restrict__ Tf,
               float* __restrict__ out2, float* __restrict__ gstats){
    __shared__ unsigned char qs[64*CK];   // 36,864 B: patches for 64 positions
    __shared__ float ssum[64], ssq[64];
    int tid = threadIdx.x;
    if (tid < 64){ ssum[tid] = 0.0f; ssq[tid] = 0.0f; }

    // ---- stage quantized 3x3 patch rows for 64 consecutive positions ----
    for (int e = tid; e < 64*CK; e += 64){
        int r  = e / CK, ck = e - r*CK;           // r = local position 0..63
        int p  = blockIdx.x*64 + r;
        int b  = p / HWHW, rem = p - b*HWHW;
        int y  = rem / HWD, xx = rem - y*HWD;
        int c  = ck / 9,  k9 = ck - c*9;
        int kh = k9 / 3,  kw = k9 - kh*3;
        int iy = y + kh - 1, ix = xx + kw - 1;
        unsigned char v = 0;                      // zero-pad -> level 0 exactly
        if ((unsigned)iy < 56u && (unsigned)ix < 56u)
            v = act1q[((size_t)(b*OO + c))*HWHW + iy*HWD + ix];
        qs[e] = v;
    }
    __syncthreads();

    int lane = tid & 31, w = tid >> 5;            // wave id 0..1
    int m    = lane & 15;
    int base = (lane < 16) ? 0 : 8;               // A-frag: lanes 16-31 carry K 16..31 (odd ck)
    v8f acc[2][4];
    #pragma unroll
    for (int mt = 0; mt < 2; ++mt)
        #pragma unroll
        for (int nt = 0; nt < 4; ++nt) acc[mt][nt] = (v8f){};

    const unsigned char* qrow0 = &qs[(w*32 +  0 + m)*CK];   // M-tile 0 of this wave
    const unsigned char* qrow1 = &qs[(w*32 + 16 + m)*CK];   // M-tile 1 of this wave
    const unsigned short* tbase = Tf + (size_t)lane*16;

    // one k-step: build two one-hot A frags, do 8 accumulating WMMAs
    auto kstep = [&](unsigned qp0, unsigned qp1, v16u (&bv)[4]){
        int a0 = (int)((qp0 >> base) & 0xFFu);
        int b0 = (int)((qp1 >> base) & 0xFFu);
        v16u av0, av1;
        #pragma unroll
        for (int j = 0; j < 16; ++j){
            av0[j]     = (a0 == j) ? (unsigned short)0x3F80 : (unsigned short)0;
            av1[j]     = (b0 == j) ? (unsigned short)0x3F80 : (unsigned short)0;
        }
        #pragma unroll
        for (int nt = 0; nt < 4; ++nt){
            acc[0][nt] = __builtin_amdgcn_wmma_f32_16x16x32_bf16(
                             false, __builtin_bit_cast(v16bf, av0),
                             false, __builtin_bit_cast(v16bf, bv[nt]),
                             (short)0, acc[0][nt], false, false);
            acc[1][nt] = __builtin_amdgcn_wmma_f32_16x16x32_bf16(
                             false, __builtin_bit_cast(v16bf, av1),
                             false, __builtin_bit_cast(v16bf, bv[nt]),
                             (short)0, acc[1][nt], false, false);
        }
    };

    v16u ba[4], bb[4];                            // ping-pong B fragments
    #pragma unroll
    for (int nt = 0; nt < 4; ++nt)
        ba[nt] = *(const v16u*)(tbase + (size_t)(nt*32)*16);

    for (int kk = 0; kk < CK/2; kk += 2){
        unsigned qq0 = *(const unsigned*)(qrow0 + kk*2);    // 4 quant indices
        unsigned qq1 = *(const unsigned*)(qrow1 + kk*2);
        #pragma unroll
        for (int nt = 0; nt < 4; ++nt)                      // loads for step kk+1
            bb[nt] = *(const v16u*)(tbase + (size_t)((kk+1)*128 + nt*32)*16);
        if (kk + 16 < CK/2)                                 // prefetch into WGP$
            __builtin_prefetch(tbase + (size_t)(kk + 16)*128*16, 0, 3);

        kstep(qq0 & 0xFFFFu, qq1 & 0xFFFFu, ba);            // step kk (uses ba)

        if (kk + 2 < CK/2){
            #pragma unroll
            for (int nt = 0; nt < 4; ++nt)                  // loads for step kk+2
                ba[nt] = *(const v16u*)(tbase + (size_t)((kk+2)*128 + nt*32)*16);
        }
        kstep(qq0 >> 16, qq1 >> 16, bb);                    // step kk+1 (uses bb)
    }

    // ---- epilogue: store D tiles, accumulate BN2 statistics ----
    int o0 = lane & 15;                           // D layout: N = lane%16
    float s[4] = {0,0,0,0}, s2[4] = {0,0,0,0};
    #pragma unroll
    for (int mt = 0; mt < 2; ++mt){
        #pragma unroll
        for (int v = 0; v < 8; ++v){
            int M = v + ((lane >> 4) << 3);       // D layout: M = vgpr + 8*(lane>=16)
            int p = blockIdx.x*64 + w*32 + mt*16 + M;
            int b = p / HWHW, rem = p - b*HWHW;
            int y = rem / HWD, xx = rem - y*HWD;
            size_t baseidx = ((size_t)b*OO)*HWHW + y*HWD + xx;
            #pragma unroll
            for (int nt = 0; nt < 4; ++nt){
                float val = -acc[mt][nt][v];
                out2[baseidx + (size_t)(nt*16 + o0)*HWHW] = val;
                s[nt] += val; s2[nt] += val*val;
            }
        }
    }
    #pragma unroll
    for (int nt = 0; nt < 4; ++nt){
        atomicAdd(&ssum[nt*16 + o0], s[nt]);
        atomicAdd(&ssq[nt*16 + o0],  s2[nt]);
    }
    __syncthreads();
    if (tid < 64){
        atomicAdd(&gstats[128 + tid], ssum[tid]);
        atomicAdd(&gstats[192 + tid], ssq[tid]);
    }
}

// ---------------- K5: BN2 + residual + ReLU + clip + quantize -> output -------
__global__ void k5_final(const float* __restrict__ out2, const float* __restrict__ gstats,
                         const float* __restrict__ s2, const float* __restrict__ b2,
                         const float* __restrict__ xres, float* __restrict__ outp){
    int idx = blockIdx.x * 256 + threadIdx.x;
    if (idx >= NPIX) return;
    int o = (idx / HWHW) & 63;
    float mean = gstats[128 + o] * (1.0f/NRED);
    float var  = gstats[192 + o] * (1.0f/NRED) - mean*mean;
    float inv  = rsqrtf(var + EPSV);
    float y = (out2[idx] - mean) * inv * s2[o] + b2[o] + xres[idx];
    y = fminf(fmaxf(y, 0.0f), 6.0f);
    outp[idx] = rintf(y * INVDELTA) * DELTA;
}

extern "C" void kernel_launch(void* const* d_in, const int* in_sizes, int n_in,
                              void* d_out, int out_size, void* d_ws, size_t ws_size,
                              hipStream_t stream){
    (void)in_sizes; (void)n_in; (void)out_size; (void)ws_size;
    const float* x    = (const float*)d_in[0];
    const float* w1   = (const float*)d_in[1];
    const float* w2   = (const float*)d_in[2];
    const float* bn1s = (const float*)d_in[3];
    const float* bn1b = (const float*)d_in[4];
    const float* bn2s = (const float*)d_in[5];
    const float* bn2b = (const float*)d_in[6];

    char* ws = (char*)d_ws;
    float*          gstats = (float*)ws;                         //      1,024 B
    float*          out1   = (float*)(ws + 1024);                // 12,845,056 B
    unsigned char*  act1q  = (unsigned char*)(ws + 12846080);    //  3,211,264 B
    float*          out2   = (float*)(ws + 16057344);            // 12,845,056 B
    unsigned short* Tf     = (unsigned short*)(ws + 28902400);   //  1,179,648 B

    hipLaunchKernelGGL(k0_zero,        dim3(1),               dim3(256), 0, stream, gstats);
    hipLaunchKernelGGL(k1_adder1,      dim3(BB*49),           dim3(256), 0, stream, x, w1, out1, gstats);
    hipLaunchKernelGGL(k2_bnq,         dim3((NPIX+255)/256),  dim3(256), 0, stream, out1, gstats, bn1s, bn1b, act1q);
    hipLaunchKernelGGL(k3_lut,         dim3((OO*CK*16+255)/256), dim3(256), 0, stream, w2, Tf);
    hipLaunchKernelGGL(k4_adder2_wmma, dim3(NRED/64),         dim3(64),  0, stream, act1q, Tf, out2, gstats);
    hipLaunchKernelGGL(k5_final,       dim3((NPIX+255)/256),  dim3(256), 0, stream, out2, gstats, bn2s, bn2b, x, (float*)d_out);
}
